// Aggregator_8796093022750
// MI455X (gfx1250) — matrix operand order repaired and checked by reading, not compiled, not run on hardware
//
#include <hip/hip_runtime.h>

typedef __attribute__((ext_vector_type(16))) _Float16 v16h;
typedef __attribute__((ext_vector_type(8)))  float    v8f;

#define N_CAT   100000
#define N_USERS 50000
#define N_REL   32
#define DIM     64
#define NEDGE   1600000
#define NNZ_CNT 1000000

// ---------------- zero init (d_out + grp_max/grp_sum in ws) ----------------
__global__ __launch_bounds__(256) void k_zero(float* __restrict__ out, int out_n,
                                              float* __restrict__ grp, int grp_n) {
  int i = blockIdx.x * blockDim.x + threadIdx.x;
  if (i < out_n) out[i] = 0.0f;
  else if (i < out_n + grp_n) grp[i - out_n] = 0.0f;
}

// ---------------- pass 1: att[e] = ||h*r||^2 * ||t*r||^2, segment max ------
__global__ __launch_bounds__(256) void k_edge_att(
    const float* __restrict__ cat_emb, const int* __restrict__ head,
    const int* __restrict__ tail, const int* __restrict__ etype,
    const float* __restrict__ weight, float* __restrict__ att,
    float* __restrict__ grp_max, int n_edges) {
  __shared__ float w_lds[N_REL * DIM];                    // 8 KB, reused by all waves
  for (int i = threadIdx.x; i < N_REL * DIM; i += blockDim.x) w_lds[i] = weight[i];
  __syncthreads();

  int lane = threadIdx.x & 31;
  int e = blockIdx.x * 8 + (threadIdx.x >> 5);            // one wave per edge
  if (e >= n_edges) return;
  int h = head[e], t = tail[e], r = etype[e] - 1;

  const float2* ch = (const float2*)(cat_emb + (size_t)h * DIM);
  const float2* ct = (const float2*)(cat_emb + (size_t)t * DIM);
  const float2* wr = (const float2*)(w_lds + r * DIM);
  float2 hv = ch[lane], tv = ct[lane], wv = wr[lane];     // 2 dims per lane
  float a0 = hv.x * wv.x, a1 = hv.y * wv.y;
  float b0 = tv.x * wv.x, b1 = tv.y * wv.y;
  float hr2 = a0 * a0 + a1 * a1;
  float tr2 = b0 * b0 + b1 * b1;
#pragma unroll
  for (int off = 16; off > 0; off >>= 1) {
    hr2 += __shfl_xor(hr2, off, 32);
    tr2 += __shfl_xor(tr2, off, 32);
  }
  if (lane == 0) {
    float a = hr2 * tr2;                                  // (||h*r||*||t*r||)^2, >= 0
    att[e] = a;
    // non-negative floats: uint bit-pattern compare == float compare
    atomicMax((unsigned int*)(grp_max + h), __float_as_uint(a));
  }
}

// ---------------- pass 2: e = exp(att - max[head]); segment sum ------------
__global__ __launch_bounds__(256) void k_edge_exp(
    const int* __restrict__ head, float* __restrict__ att,
    const float* __restrict__ grp_max, float* __restrict__ grp_sum, int n_edges) {
  int e = blockIdx.x * blockDim.x + threadIdx.x;
  if (e >= n_edges) return;
  int h = head[e];
  float ev = __expf(att[e] - grp_max[h]);
  att[e] = ev;
  atomicAdd(grp_sum + h, ev);
}

// ---------------- pass 3: category_agg[head] += w * (t*r) ------------------
__global__ __launch_bounds__(256) void k_edge_agg(
    const float* __restrict__ cat_emb, const int* __restrict__ head,
    const int* __restrict__ tail, const int* __restrict__ etype,
    const float* __restrict__ weight, const float* __restrict__ att,
    const float* __restrict__ grp_sum, float* __restrict__ cat_agg, int n_edges) {
  __shared__ float w_lds[N_REL * DIM];
  for (int i = threadIdx.x; i < N_REL * DIM; i += blockDim.x) w_lds[i] = weight[i];
  __syncthreads();

  int lane = threadIdx.x & 31;
  int e = blockIdx.x * 8 + (threadIdx.x >> 5);
  if (e >= n_edges) return;
  int h = head[e], t = tail[e], r = etype[e] - 1;
  float w = att[e] / grp_sum[h];

  const float2* ct = (const float2*)(cat_emb + (size_t)t * DIM);
  const float2* wr = (const float2*)(w_lds + r * DIM);
  float2 tv = ct[lane], wv = wr[lane];
  float* dst = cat_agg + (size_t)h * DIM + lane * 2;
  atomicAdd(dst,     w * tv.x * wv.x);
  atomicAdd(dst + 1, w * tv.y * wv.y);
}

// ---------------- COO SpMM: user_agg[row] += val * cat_emb[col] ------------
__global__ __launch_bounds__(256) void k_spmm(
    const float* __restrict__ cat_emb, const int* __restrict__ rows,
    const int* __restrict__ cols, const float* __restrict__ vals,
    float* __restrict__ user_agg, int nnz) {
  int lane = threadIdx.x & 31;
  int i = blockIdx.x * 8 + (threadIdx.x >> 5);            // one wave per nnz
  if (i >= nnz) return;
  int rr = rows[i], cc = cols[i];
  float v = vals[i];
  const float2* cp = (const float2*)(cat_emb + (size_t)cc * DIM);
  float2 cv = cp[lane];
  float* dst = user_agg + (size_t)rr * DIM + lane * 2;
  atomicAdd(dst,     v * cv.x);
  atomicAdd(dst + 1, v * cv.y);
}

// ---------------- fused user GEMM via WMMA ---------------------------------
// user_agg[u] *= 1 + (softmax(U @ W^T) @ W)[u], one wave per 16-user tile.
__global__ __launch_bounds__(256) void k_user_gemm(
    const float* __restrict__ user_emb, const float* __restrict__ weight,
    float* __restrict__ user_agg, int n_users) {
  // f16 weight, row-major [n=rel][k=dim]: GEMM1 B fragment (B[k][n] = W[n][k])
  __shared__ __align__(32) _Float16 w16[N_REL * DIM];
  // f16 weight, transposed [n=dim][k=rel]: GEMM2 B fragment (B[k][n] = W[k][n])
  __shared__ __align__(32) _Float16 wT16[DIM * N_REL];
  // per-wave pre-softmax logits S[m][n] in f32; row stride 36 -> 16B align,
  // 36*r mod 64 enumerates multiples of 4 -> conflict-free b128 reads
  __shared__ __align__(16) float sf[8][16][36];

  for (int i = threadIdx.x; i < N_REL * DIM; i += blockDim.x) {
    int r = i >> 6, d = i & 63;                           // i = r*64 + d
    float w = weight[i];
    w16[i] = (_Float16)w;
    wT16[d * N_REL + r] = (_Float16)w;
  }
  __syncthreads();

  int lane = threadIdx.x & 31;
  int wid  = threadIdx.x >> 5;
  int row  = lane & 15;
  int hi   = lane >> 4;
  int tile = blockIdx.x * 8 + wid;
  int u0   = tile * 16;
  bool valid = (u0 + 16) <= n_users;                      // 50000 % 16 == 0

  int urow = u0 + row;
  if (urow > n_users - 1) urow = n_users - 1;             // clamp: branch-free OOB
  const float* ubase = user_emb + (size_t)urow * DIM;

  // ---- GEMM1: S_pre = U(16x64) @ W^T(64x32), two 16x16 N-tiles -----------
  v8f c0 = {}, c1 = {};
#pragma unroll
  for (int k0 = 0; k0 < DIM; k0 += 32) {
    // A striping: K = k0 + hi*16 + 0..15 -> 4x b128 loads
    const float4* pa0 = (const float4*)(ubase + k0 + hi * 16);
    const float4* pa1 = (const float4*)(ubase + k0 + hi * 16 + 8);
    float4 x0 = pa0[0], x1 = pa0[1];
    float4 y0 = pa1[0], y1 = pa1[1];
    v16h a;
    a[0]  = (_Float16)x0.x; a[1]  = (_Float16)x0.y;
    a[2]  = (_Float16)x0.z; a[3]  = (_Float16)x0.w;
    a[4]  = (_Float16)x1.x; a[5]  = (_Float16)x1.y;
    a[6]  = (_Float16)x1.z; a[7]  = (_Float16)x1.w;
    a[8]  = (_Float16)y0.x; a[9]  = (_Float16)y0.y;
    a[10] = (_Float16)y0.z; a[11] = (_Float16)y0.w;
    a[12] = (_Float16)y1.x; a[13] = (_Float16)y1.y;
    a[14] = (_Float16)y1.z; a[15] = (_Float16)y1.w;
    // B striping: K = k0 + hi*16 + 0..15, contiguous f16 -> one 32B LDS load
    v16h b0 = *(const v16h*)(w16 + row * DIM + k0 + hi * 16);        // n = row
    v16h b1 = *(const v16h*)(w16 + (16 + row) * DIM + k0 + hi * 16); // n = 16+row
    c0 = __builtin_amdgcn_wmma_f32_16x16x32_f16(false, a, false, b0, (short)0, c0, false, false);
    c1 = __builtin_amdgcn_wmma_f32_16x16x32_f16(false, a, false, b1, (short)0, c1, false, false);
  }

  // ---- C-layout -> LDS (f32 logits): reg r holds M=r+8*hi, N=row/16+row --
#pragma unroll
  for (int r = 0; r < 8; ++r) {
    sf[wid][r + 8 * hi][row]      = c0[r];
    sf[wid][r + 8 * hi][16 + row] = c1[r];
  }
  __syncthreads();

  // ---- reload in A striping; softmax needs only ONE cross-lane xor-16 ----
  // lane (row,hi) holds K = hi*16 + 0..15 of row `row`;
  // partner lane (row, 1-hi) holds the complementary 16 values.
  const float* srow = &sf[wid][row][0];
  float4 q0 = *(const float4*)(srow + hi * 16);
  float4 q1 = *(const float4*)(srow + hi * 16 + 4);
  float4 q2 = *(const float4*)(srow + hi * 16 + 8);
  float4 q3 = *(const float4*)(srow + hi * 16 + 12);
  float v[16] = {q0.x, q0.y, q0.z, q0.w, q1.x, q1.y, q1.z, q1.w,
                 q2.x, q2.y, q2.z, q2.w, q3.x, q3.y, q3.z, q3.w};
  float m = v[0];
#pragma unroll
  for (int i = 1; i < 16; ++i) m = fmaxf(m, v[i]);
  m = fmaxf(m, __shfl_xor(m, 16, 32));                    // combine half-rows
  float ev[16], s = 0.0f;
#pragma unroll
  for (int i = 0; i < 16; ++i) { ev[i] = __expf(v[i] - m); s += ev[i]; }
  s += __shfl_xor(s, 16, 32);
  float inv = 1.0f / s;
  v16h a2;
#pragma unroll
  for (int i = 0; i < 16; ++i) a2[i] = (_Float16)(ev[i] * inv);

  // ---- GEMM2: P = S(16x32) @ W(32x64), four 16x16 N-tiles, K=32 ----------
  v8f p[4];
#pragma unroll
  for (int nt = 0; nt < 4; ++nt) {
    // B[k][n] = W[k][n]; n = nt*16 + row, K = hi*16 + 0..15 contiguous in wT16
    v16h b2 = *(const v16h*)(wT16 + (nt * 16 + row) * N_REL + hi * 16);
    v8f z = {};
    p[nt] = __builtin_amdgcn_wmma_f32_16x16x32_f16(false, a2, false, b2, (short)0, z, false, false);
  }

  // ---- in-place: user_agg[u][d] *= 1 + P[u][d] ---------------------------
  if (valid) {
#pragma unroll
    for (int nt = 0; nt < 4; ++nt) {
#pragma unroll
      for (int r = 0; r < 8; ++r) {
        size_t idx = (size_t)(u0 + r + 8 * hi) * DIM + nt * 16 + row;
        user_agg[idx] *= (1.0f + p[nt][r]);
      }
    }
  }
}

extern "C" void kernel_launch(void* const* d_in, const int* in_sizes, int n_in,
                              void* d_out, int out_size, void* d_ws, size_t ws_size,
                              hipStream_t stream) {
  const float* cat_emb  = (const float*)d_in[0];
  const float* user_emb = (const float*)d_in[1];
  const int*   edge_idx = (const int*)d_in[2];
  const int*   etype    = (const int*)d_in[3];
  const int*   im_rows  = (const int*)d_in[4];
  const int*   im_cols  = (const int*)d_in[5];
  const float* im_vals  = (const float*)d_in[6];
  const float* weight   = (const float*)d_in[7];

  float* out      = (float*)d_out;
  float* cat_agg  = out;                                  // [N_CAT, 64]
  float* user_agg = out + (size_t)N_CAT * DIM;            // [N_USERS, 64]

  float* ws      = (float*)d_ws;
  float* att     = ws;                                    // [E]
  float* grp_max = ws + NEDGE;                            // [N_CAT]
  float* grp_sum = grp_max + N_CAT;                       // [N_CAT]

  const int* head = edge_idx;
  const int* tail = edge_idx + NEDGE;

  int ztotal = out_size + 2 * N_CAT;
  k_zero<<<(ztotal + 255) / 256, 256, 0, stream>>>(out, out_size, grp_max, 2 * N_CAT);
  k_edge_att<<<(NEDGE + 7) / 8, 256, 0, stream>>>(cat_emb, head, tail, etype, weight,
                                                  att, grp_max, NEDGE);
  k_edge_exp<<<(NEDGE + 255) / 256, 256, 0, stream>>>(head, att, grp_max, grp_sum, NEDGE);
  k_edge_agg<<<(NEDGE + 7) / 8, 256, 0, stream>>>(cat_emb, head, tail, etype, weight,
                                                  att, grp_sum, cat_agg, NEDGE);
  k_spmm<<<(NNZ_CNT + 7) / 8, 256, 0, stream>>>(cat_emb, im_rows, im_cols, im_vals,
                                                user_agg, NNZ_CNT);
  int tiles = (N_USERS + 15) / 16;
  k_user_gemm<<<(tiles + 7) / 8, 256, 0, stream>>>(user_emb, weight, user_agg, N_USERS);
}
